// MeshDecoder_57415122812965
// MI455X (gfx1250) — matrix operand order repaired and checked by reading, not compiled, not run on hardware
//
#include <hip/hip_runtime.h>

typedef __attribute__((ext_vector_type(16))) __bf16 v16bf;
typedef __attribute__((ext_vector_type(8)))  float  v8f;

__device__ __forceinline__ unsigned short f2bf(float f) {
    union { float f; unsigned int u; } x; x.f = f;
    unsigned int u = x.u;
    unsigned int r = u + 0x7FFFu + ((u >> 16) & 1u);   // round-to-nearest-even
    return (unsigned short)(r >> 16);
}

__device__ __forceinline__ unsigned int pack2bf(float lo, float hi) {
    return (unsigned int)f2bf(lo) | ((unsigned int)f2bf(hi) << 16);
}

// ---------------------------------------------------------------------------
// Pack W (Cout, Cin, 5) fp32  ->  Wp[CoutP][5][CinP] bf16 (zero padded)
// ---------------------------------------------------------------------------
__global__ void pack_w(const float* __restrict__ W, unsigned short* __restrict__ Wp,
                       int Cout, int Cin, int CinP, int total) {
    int i = blockIdx.x * 256 + threadIdx.x;
    if (i >= total) return;
    int c = i % CinP;
    int s = (i / CinP) % 5;
    int o = i / (CinP * 5);
    float v = 0.0f;
    if (o < Cout && c < Cin) v = W[((size_t)o * Cin + c) * 5 + s];
    Wp[i] = f2bf(v);
}

// ---------------------------------------------------------------------------
// Transpose channel-major (B,C,E) -> edge-major (B,E,C)
// ---------------------------------------------------------------------------
__global__ void transpose_cm_em(const float* __restrict__ in, float* __restrict__ out,
                                int C, int E) {
    __shared__ float tile[32][33];
    int b  = blockIdx.z;
    int e0 = blockIdx.x * 32;
    int c0 = blockIdx.y * 32;
    int tx = threadIdx.x & 31, ty = threadIdx.x >> 5;    // 32 x 8
    for (int j = 0; j < 32; j += 8) {
        int c = c0 + ty + j;
        if (c < C) tile[ty + j][tx] = in[((size_t)b * C + c) * E + e0 + tx];
    }
    __syncthreads();
    for (int j = 0; j < 32; j += 8) {
        int e = e0 + ty + j;
        int c = c0 + tx;
        if (c < C) out[((size_t)b * E + e) * C + c] = tile[tx][ty + j];
    }
}

// Transpose edge-major (B,E,C) -> channel-major (B,C,E)
__global__ void transpose_em_cm(const float* __restrict__ in, float* __restrict__ out,
                                int C, int E) {
    __shared__ float tile[32][33];
    int b  = blockIdx.z;
    int e0 = blockIdx.x * 32;
    int c0 = blockIdx.y * 32;
    int tx = threadIdx.x & 31, ty = threadIdx.x >> 5;
    for (int j = 0; j < 32; j += 8) {
        int e = e0 + ty + j;
        int c = c0 + tx;
        if (c < C) tile[tx][ty + j] = in[((size_t)b * E + e) * C + c];
    }
    __syncthreads();
    for (int j = 0; j < 32; j += 8) {
        int c = c0 + ty + j;
        if (c < C) out[((size_t)b * C + c) * E + e0 + tx] = tile[ty + j][tx];
    }
}

// ---------------------------------------------------------------------------
// Y[b,e2,c] = U[b, up[e2], c] + skip[b,c,e2]   (skip is channel-major input)
// C is a multiple of 32 here (128/64/32).
// ---------------------------------------------------------------------------
__global__ void unpool_skip(const float* __restrict__ U, const int* __restrict__ up,
                            const float* __restrict__ skip, float* __restrict__ Y,
                            int E1, int E2, int C) {
    __shared__ float tile[32][33];
    int b  = blockIdx.z;
    int e0 = blockIdx.x * 32;
    int c0 = blockIdx.y * 32;
    int tx = threadIdx.x & 31, ty = threadIdx.x >> 5;
    for (int j = 0; j < 32; j += 8) {
        int c = c0 + ty + j;
        tile[ty + j][tx] = skip[((size_t)b * C + c) * E2 + e0 + tx];
    }
    __syncthreads();
    for (int j = 0; j < 32; j += 8) {
        int e = e0 + ty + j;
        int c = c0 + tx;
        int src = up[e];
        Y[((size_t)b * E2 + e) * C + c] =
            U[((size_t)b * E1 + src) * C + c] + tile[tx][ty + j];
    }
}

// ---------------------------------------------------------------------------
// Mesh-conv GEMM with bf16 WMMA.
//   x:    (B,E,Cin) edge-major fp32
//   gemm: (E,4) neighbor indices
//   Wp:   bf16 [CoutP][5][CinP]   (K ordering: k = s*CinP + c)
//   out:  (B,E,Cout) edge-major fp32 (bias fused)
// Block = 256 threads (8 waves); tile = 128 edges; wave w -> 16-edge N-tile;
// each wave iterates MT M-tiles (MT = CoutP/16).
// ---------------------------------------------------------------------------
template <int MT>
__global__ __launch_bounds__(256)
void conv_bf16(const float* __restrict__ x, const int* __restrict__ gemm,
               const unsigned short* __restrict__ Wp, const float* __restrict__ bias,
               float* __restrict__ out, int E, int Cin, int CinP, int Cout) {
    __shared__ unsigned short Gs[5][128][32];   // 40 KB feature tile (bf16)
    __shared__ int idxs[128][4];

    const int t    = threadIdx.x;
    const int b    = blockIdx.y;
    const int tile = blockIdx.x * 128;
    const size_t xb = (size_t)b * E * Cin;

    for (int i = t; i < 512; i += 256)
        ((int*)idxs)[i] = gemm[(size_t)tile * 4 + i];

    v8f acc[MT] = {};

    const int lane = t & 31;
    const int wave = t >> 5;
    const int n0   = wave * 16;
    const int lrow = lane & 15;
    const int kbA  = (lane < 16) ? 0 : 8;    // A-matrix K split  (K{0-7,16-23} / K{8-15,24-31})
    const int kbB  = (lane < 16) ? 0 : 8;    // B-matrix K split  (K{0-7,16-23} / K{8-15,24-31})
    const int obD  = (lane < 16) ? 0 : 8;    // D-matrix M offset

    // gather-phase thread mapping: each thread owns a channel PAIR
    const int cl2 = (t & 15) * 2;            // channel-pair offset within chunk
    const int nr  = t >> 4;                  // 0..15 edge row within pass

    __syncthreads();

    for (int cc = 0; cc < CinP; cc += 32) {
        // ---- gather + feature build: G[5][128 edges][32 channels] (bf16) ----
        {
            const int  c  = cc + cl2;
            const bool cv = (c < Cin);
            for (int p = 0; p < 8; ++p) {
                int n = p * 16 + nr;
                int e = tile + n;
                float2 sf = {0.f, 0.f}, g1 = {0.f, 0.f}, g2 = {0.f, 0.f},
                       g3 = {0.f, 0.f}, g4 = {0.f, 0.f};
                if (cv) {
                    sf = *(const float2*)&x[xb + (size_t)e * Cin + c];
                    g1 = *(const float2*)&x[xb + (size_t)idxs[n][0] * Cin + c];
                    g2 = *(const float2*)&x[xb + (size_t)idxs[n][1] * Cin + c];
                    g3 = *(const float2*)&x[xb + (size_t)idxs[n][2] * Cin + c];
                    g4 = *(const float2*)&x[xb + (size_t)idxs[n][3] * Cin + c];
                }
                *(unsigned int*)&Gs[0][n][cl2] = pack2bf(sf.x, sf.y);
                *(unsigned int*)&Gs[1][n][cl2] = pack2bf(g1.x + g3.x, g1.y + g3.y);
                *(unsigned int*)&Gs[2][n][cl2] = pack2bf(g2.x + g4.x, g2.y + g4.y);
                *(unsigned int*)&Gs[3][n][cl2] = pack2bf(fabsf(g1.x - g3.x), fabsf(g1.y - g3.y));
                *(unsigned int*)&Gs[4][n][cl2] = pack2bf(fabsf(g2.x - g4.x), fabsf(g2.y - g4.y));
            }
        }
        __syncthreads();

        // ---- 5 WMMA K-steps (one per feature plane) over this channel chunk ----
        #pragma unroll
        for (int s = 0; s < 5; ++s) {
            union { v16bf v; uint4 q[2]; } Bf;
            const unsigned short* gr = &Gs[s][n0 + lrow][kbB];
            Bf.q[0] = *(const uint4*)gr;
            Bf.q[1] = *(const uint4*)(gr + 16);
            #pragma unroll
            for (int mt = 0; mt < MT; ++mt) {
                const unsigned short* wr =
                    Wp + ((size_t)(mt * 16 + lrow) * 5 + s) * CinP + cc + kbA;
                union { v16bf v; uint4 q[2]; } Af;
                Af.q[0] = *(const uint4*)wr;
                Af.q[1] = *(const uint4*)(wr + 16);
                acc[mt] = __builtin_amdgcn_wmma_f32_16x16x32_bf16(
                    false, Af.v, false, Bf.v, (short)0, acc[mt], false, false);
            }
        }
        __syncthreads();
    }

    // ---- epilogue: D layout -> edge-major store, bias fused ----
    const int eg = tile + n0 + lrow;
    #pragma unroll
    for (int mt = 0; mt < MT; ++mt) {
        int o0 = mt * 16 + obD;
        if (o0 < Cout) {
            float* po = out + ((size_t)b * E + eg) * Cout + o0;
            float4 lo, hi;
            lo.x = acc[mt][0] + bias[o0 + 0];
            lo.y = acc[mt][1] + bias[o0 + 1];
            lo.z = acc[mt][2] + bias[o0 + 2];
            lo.w = acc[mt][3] + bias[o0 + 3];
            hi.x = acc[mt][4] + bias[o0 + 4];
            hi.y = acc[mt][5] + bias[o0 + 5];
            hi.z = acc[mt][6] + bias[o0 + 6];
            hi.w = acc[mt][7] + bias[o0 + 7];
            *(float4*)po       = lo;
            *(float4*)(po + 4) = hi;
        }
    }
}

// ---------------------------------------------------------------------------
// Instance-norm: deterministic two-pass reduction over E per (b,c), then
// in-place normalize + ReLU.  C is a power of two (128/64/32/8).
// ---------------------------------------------------------------------------
__global__ void inorm_partials(const float* __restrict__ x, float* __restrict__ partials,
                               int E, int C, int RB) {
    int b = blockIdx.y, blk = blockIdx.x;
    int chunk = E / RB;
    int rows  = 256 / C;
    int c = threadIdx.x % C;
    int r = threadIdx.x / C;
    float s1 = 0.f, s2 = 0.f;
    int e_end = blk * chunk + chunk;
    for (int e = blk * chunk + r; e < e_end; e += rows) {
        float v = x[((size_t)b * E + e) * C + c];
        s1 += v; s2 += v * v;
    }
    __shared__ float sh1[256], sh2[256];
    sh1[threadIdx.x] = s1; sh2[threadIdx.x] = s2;
    __syncthreads();
    for (int half = rows >> 1; half >= 1; half >>= 1) {
        if (r < half) {
            sh1[threadIdx.x] += sh1[threadIdx.x + half * C];
            sh2[threadIdx.x] += sh2[threadIdx.x + half * C];
        }
        __syncthreads();
    }
    if (r == 0) {
        size_t o = (((size_t)b * RB + blk) * C + c) * 2;
        partials[o]     = sh1[c];
        partials[o + 1] = sh2[c];
    }
}

__global__ void inorm_finalize(const float* __restrict__ partials, float* __restrict__ stats,
                               int E, int C, int RB) {
    int b = blockIdx.x;
    int c = threadIdx.x;
    if (c >= C) return;
    float s1 = 0.f, s2 = 0.f;
    for (int k = 0; k < RB; ++k) {
        size_t o = (((size_t)b * RB + k) * C + c) * 2;
        s1 += partials[o];
        s2 += partials[o + 1];
    }
    float m = s1 / (float)E;
    float v = s2 / (float)E - m * m;
    stats[((size_t)b * C + c) * 2]     = m;
    stats[((size_t)b * C + c) * 2 + 1] = rsqrtf(v + 1e-5f);
}

__global__ void norm_relu(float* __restrict__ x, const float* __restrict__ stats,
                          int E, int C, int total) {
    int i = blockIdx.x * 256 + threadIdx.x;
    if (i >= total) return;
    int c = i % C;
    int b = i / (E * C);
    float m  = stats[((size_t)b * C + c) * 2];
    float rs = stats[((size_t)b * C + c) * 2 + 1];
    float v = (x[i] - m) * rs;
    x[i] = v > 0.f ? v : 0.f;
}

// ---------------------------------------------------------------------------
// Driver
// ---------------------------------------------------------------------------
extern "C" void kernel_launch(void* const* d_in, const int* in_sizes, int n_in,
                              void* d_out, int out_size, void* d_ws, size_t ws_size,
                              hipStream_t stream) {
    const float* fe = (const float*)d_in[0];
    const float* skip[3]   = {(const float*)d_in[1], (const float*)d_in[2], (const float*)d_in[3]};
    const int*   gemm[4]   = {(const int*)d_in[4], (const int*)d_in[5], (const int*)d_in[6], (const int*)d_in[7]};
    const int*   unpool[3] = {(const int*)d_in[8], (const int*)d_in[9], (const int*)d_in[10]};
    const float* W[8]; const float* bias[8];
    for (int k = 0; k < 8; ++k) {            // up0,c0,up1,c1,up2,c2,upf,cf
        W[k]    = (const float*)d_in[11 + 2 * k];
        bias[k] = (const float*)d_in[12 + 2 * k];
    }

    char* ws = (char*)d_ws;
    float* A = (float*)(ws);                               // 33,554,432 B arena
    float* U = (float*)(ws + 33554432ull);                 // 16,777,216 B arena
    float* Y = (float*)(ws + 50331648ull);                 // 33,554,432 B arena
    char*  WpArena = ws + 83886080ull;                     // ~0.64 MB packed weights
    float* P = (float*)(ws + 84934656ull);                 // partials
    float* S = (float*)(ws + 85458944ull);                 // stats

    // weight dims: Cout, CoutP, Cin, CinP
    const int wCout[8]  = {128, 128, 64, 64, 32, 32,  8,  8};
    const int wCoutP[8] = {128, 128, 64, 64, 32, 32, 16, 16};
    const int wCin[8]   = {256, 128, 128, 64, 64, 32, 32,  8};
    const int wCinP[8]  = {256, 128, 128, 64, 64, 32, 32, 32};
    const size_t wpOff[8] = {0, 327680, 491520, 573440, 614400, 634880, 645120, 650240};
    unsigned short* Wp[8];
    for (int k = 0; k < 8; ++k) {
        Wp[k] = (unsigned short*)(WpArena + wpOff[k]);
        int total = wCoutP[k] * 5 * wCinP[k];
        pack_w<<<(total + 255) / 256, 256, 0, stream>>>(W[k], Wp[k], wCout[k], wCin[k], wCinP[k], total);
    }

    const int RB = 128;
    auto run_inorm = [&](float* X, int E, int C) {
        inorm_partials<<<dim3(RB, 2), 256, 0, stream>>>(X, P, E, C, RB);
        inorm_finalize<<<dim3(2), 128, 0, stream>>>(P, S, E, C, RB);
        int total = 2 * E * C;
        norm_relu<<<(total + 255) / 256, 256, 0, stream>>>(X, S, E, C, total);
    };

    // fe (2,256,16384) channel-major -> edge-major arena A
    transpose_cm_em<<<dim3(16384 / 32, 256 / 32, 2), 256, 0, stream>>>(fe, A, 256, 16384);

    // ---- level 0: 256 -> 128, E 16384 -> 32768 ----
    conv_bf16<8><<<dim3(16384 / 128, 2), 256, 0, stream>>>(A, gemm[0], Wp[0], bias[0], U, 16384, 256, 256, 128);
    unpool_skip<<<dim3(32768 / 32, 128 / 32, 2), 256, 0, stream>>>(U, unpool[0], skip[0], Y, 16384, 32768, 128);
    conv_bf16<8><<<dim3(32768 / 128, 2), 256, 0, stream>>>(Y, gemm[1], Wp[1], bias[1], A, 32768, 128, 128, 128);
    run_inorm(A, 32768, 128);

    // ---- level 1: 128 -> 64, E 32768 -> 65536 ----
    conv_bf16<4><<<dim3(32768 / 128, 2), 256, 0, stream>>>(A, gemm[1], Wp[2], bias[2], U, 32768, 128, 128, 64);
    unpool_skip<<<dim3(65536 / 32, 64 / 32, 2), 256, 0, stream>>>(U, unpool[1], skip[1], Y, 32768, 65536, 64);
    conv_bf16<4><<<dim3(65536 / 128, 2), 256, 0, stream>>>(Y, gemm[2], Wp[3], bias[3], A, 65536, 64, 64, 64);
    run_inorm(A, 65536, 64);

    // ---- level 2: 64 -> 32, E 65536 -> 131072 ----
    conv_bf16<2><<<dim3(65536 / 128, 2), 256, 0, stream>>>(A, gemm[2], Wp[4], bias[4], U, 65536, 64, 64, 32);
    unpool_skip<<<dim3(131072 / 32, 32 / 32, 2), 256, 0, stream>>>(U, unpool[2], skip[2], Y, 65536, 131072, 32);
    conv_bf16<2><<<dim3(131072 / 128, 2), 256, 0, stream>>>(Y, gemm[3], Wp[5], bias[5], A, 131072, 32, 32, 32);
    run_inorm(A, 131072, 32);

    // ---- final: 32 -> 8 -> 8, E 131072 ----
    conv_bf16<1><<<dim3(131072 / 128, 2), 256, 0, stream>>>(A, gemm[3], Wp[6], bias[6], U, 131072, 32, 32, 8);
    conv_bf16<1><<<dim3(131072 / 128, 2), 256, 0, stream>>>(U, gemm[3], Wp[7], bias[7], Y, 131072, 8, 32, 8);
    run_inorm(Y, 131072, 8);

    // edge-major (2,131072,8) -> output (2,8,131072)
    transpose_em_cm<<<dim3(131072 / 32, 1, 2), 256, 0, stream>>>(Y, (float*)d_out, 8, 131072);
}
